// AdaptiveDiffAttention_79250736546286
// MI455X (gfx1250) — matrix operand order repaired and checked by reading, not compiled, not run on hardware
//
#include <hip/hip_runtime.h>
#include <hip/hip_bf16.h>
#include <stdint.h>

// Problem constants (match reference)
#define B_SZ    16384
#define DIM     2048
#define SD      1024
#define NHEADS  4
#define HD      256
#define LH      256
#define TOKROWS (B_SZ * 2)          // 32768 token rows of width SD
#define SCALE_F 0.0625f             // HD^-0.5 = 1/16

typedef __attribute__((ext_vector_type(16))) __bf16 bf16x16;
typedef __attribute__((ext_vector_type(8)))  float  f32x8;

// ---------- scalar helpers ----------
static __device__ __forceinline__ unsigned short f2bf(float f) {
  union { float f; unsigned u; } v; v.f = f;
  unsigned u = v.u;
  unsigned r = (u + 0x7FFFu + ((u >> 16) & 1u)) >> 16;   // round-to-nearest-even
  return (unsigned short)r;
}
static __device__ __forceinline__ float bf2f(unsigned short s) {
  union { unsigned u; float f; } v; v.u = ((unsigned)s) << 16;
  return v.f;
}
static __device__ __forceinline__ float wave_sum(float v) {
#pragma unroll
  for (int off = 16; off > 0; off >>= 1) v += __shfl_xor(v, off, 32);
  return v;
}

// ---------- 1) f32 -> bf16 bulk convert (4 elems / thread) ----------
__global__ void __launch_bounds__(256) cvt_f32_bf16(const float* __restrict__ in,
                                                    unsigned short* __restrict__ out,
                                                    size_t n4) {
  size_t i = (size_t)blockIdx.x * blockDim.x + threadIdx.x;
  if (i >= n4) return;
  float4 f = ((const float4*)in)[i];
  unsigned long long v = (unsigned long long)f2bf(f.x)
                       | ((unsigned long long)f2bf(f.y) << 16)
                       | ((unsigned long long)f2bf(f.z) << 32)
                       | ((unsigned long long)f2bf(f.w) << 48);
  ((unsigned long long*)out)[i] = v;
}

// ---------- 2) pack W[K,N] (f32, row-major, y = x@W) into WMMA B-fragment order ----------
// Layout: [nt][kt][lane][16 bf16], h = lane/16: e<8: K = kt*32 + 8h + e ; e>=8: K = kt*32 + 16 + 8h + (e-8)
__global__ void __launch_bounds__(256) pack_weight(const float* __restrict__ W,
                                                   unsigned short* __restrict__ Wp,
                                                   int K, int N) {
  int idx = blockIdx.x * blockDim.x + threadIdx.x;
  if (idx >= K * N) return;
  int e     = idx & 15;
  int lane  = (idx >> 4) & 31;
  int Ktiles = K >> 5;
  int kt    = (idx >> 9) % Ktiles;
  int nt    = idx / (Ktiles << 9);
  int col   = (nt << 4) + (lane & 15);
  int k     = (kt << 5) + ((e >> 3) << 4) + ((lane >> 4) << 3) + (e & 7);
  Wp[idx] = f2bf(W[(size_t)k * N + col]);
}

// ---------- 3) generic bf16 WMMA GEMM: out = A[M,K] @ W[K,N] + bias (+res) (ReLU?) ----------
// Each wave: 64x32 macro-tile = 4x2 fragments; K-step = 32 -> 8 v_wmma per step.
__global__ void __launch_bounds__(256) gemm_bf16_wmma(
    const unsigned short* __restrict__ A,    // [M,K] bf16 row-major
    const unsigned short* __restrict__ Wp,   // packed fragments (see pack_weight)
    const float* __restrict__ bias,          // [N]
    const float* __restrict__ residual,      // [M,N] f32 or nullptr
    void* __restrict__ out,                  // [M,N] f32 or bf16
    int M, int N, int K, int out_bf16, int do_relu)
{
  const int lane = threadIdx.x & 31;
  const int wid  = (int)((blockIdx.x * blockDim.x + threadIdx.x) >> 5);
  const int nGroups = N >> 5;
  const int mGroups = M >> 6;
  if (wid >= mGroups * nGroups) return;
  const int mg = wid / nGroups, ng = wid % nGroups;
  const int row0 = mg << 6, col0 = ng << 5;
  const int lrow = lane & 15, lhi = lane >> 4;
  const int Ktiles = K >> 5;

  f32x8 acc[4][2];
#pragma unroll
  for (int nt = 0; nt < 2; ++nt) {
    float bv = bias[col0 + nt * 16 + lrow];
#pragma unroll
    for (int mt = 0; mt < 4; ++mt)
#pragma unroll
      for (int r = 0; r < 8; ++r) acc[mt][nt][r] = bv;
  }

  const unsigned short* aBase = A + (size_t)(row0 + lrow) * K + lhi * 8;
  const unsigned short* bBase = Wp + ((size_t)(ng * 2) * Ktiles * 32) * 16 + (size_t)lane * 16;

  for (int kt = 0; kt < Ktiles; ++kt) {
    union { uint4 u[2]; bf16x16 v; } a[4], b[2];
    const int ka = kt << 5;
#pragma unroll
    for (int mt = 0; mt < 4; ++mt) {
      const unsigned short* p = aBase + (size_t)mt * 16 * K + ka;
      a[mt].u[0] = *(const uint4*)(p);        // K = ka + lhi*8 + {0..7}
      a[mt].u[1] = *(const uint4*)(p + 16);   // K = ka + 16 + lhi*8 + {0..7}
    }
#pragma unroll
    for (int nt = 0; nt < 2; ++nt) {
      const unsigned short* p = bBase + ((size_t)nt * Ktiles + kt) * 512;
      b[nt].u[0] = *(const uint4*)(p);
      b[nt].u[1] = *(const uint4*)(p + 8);
    }
#pragma unroll
    for (int mt = 0; mt < 4; ++mt)
#pragma unroll
      for (int nt = 0; nt < 2; ++nt)
        acc[mt][nt] = __builtin_amdgcn_wmma_f32_16x16x32_bf16(
            false, a[mt].v, false, b[nt].v, (short)0, acc[mt][nt], false, false);
  }

#pragma unroll
  for (int mt = 0; mt < 4; ++mt)
#pragma unroll
    for (int nt = 0; nt < 2; ++nt) {
      const int col = col0 + nt * 16 + lrow;
#pragma unroll
      for (int r = 0; r < 8; ++r) {
        const int row = row0 + mt * 16 + lhi * 8 + r;
        float v = acc[mt][nt][r];
        if (residual) v += residual[(size_t)row * N + col];
        if (do_relu)  v = v > 0.f ? v : 0.f;
        if (out_bf16) ((unsigned short*)out)[(size_t)row * N + col] = f2bf(v);
        else          ((float*)out)[(size_t)row * N + col] = v;
      }
    }
}

// ---------- 4) lambda finish: lam[b] = sigmoid(h[b,:]·L2 + b2), one wave / row ----------
__global__ void __launch_bounds__(256) lambda_final(const unsigned short* __restrict__ h,
                                                    const float* __restrict__ L2,
                                                    const float* __restrict__ L2b,
                                                    float* __restrict__ lam, int Bn) {
  const int lane = threadIdx.x & 31;
  const int wid  = (int)((blockIdx.x * blockDim.x + threadIdx.x) >> 5);
  if (wid >= Bn) return;
  uint4 u = *(const uint4*)(h + (size_t)wid * LH + lane * 8);
  const unsigned short* s = (const unsigned short*)&u;
  float acc = 0.f;
#pragma unroll
  for (int j = 0; j < 8; ++j) acc += bf2f(s[j]) * L2[lane * 8 + j];
  acc = wave_sum(acc);
  if (lane == 0) lam[wid] = 1.f / (1.f + __expf(-(acc + L2b[0])));
}

// ---------- 5) differential attention (seq=2), one wave per (b,h) ----------
__global__ void __launch_bounds__(256) diff_attn_kernel(
    const unsigned short* __restrict__ Q1, const unsigned short* __restrict__ K1,
    const unsigned short* __restrict__ Q2, const unsigned short* __restrict__ K2,
    const unsigned short* __restrict__ V,  const float* __restrict__ lam,
    unsigned short* __restrict__ Aout)
{
  const int lane = threadIdx.x & 31;
  const int wid  = (int)((blockIdx.x * blockDim.x + threadIdx.x) >> 5);
  if (wid >= B_SZ * NHEADS) return;
  const int b = wid >> 2, h = wid & 3;
  const size_t r0 = ((size_t)b * 2) * SD + (size_t)h * HD + (size_t)lane * 8;
  const size_t r1 = r0 + SD;

  float q1a[8], q1b[8], k1a[8], k1b[8], q2a[8], q2b[8], k2a[8], k2b[8], va[8], vb[8];
  auto ld8 = [&](const unsigned short* p, size_t off, float* dst) {
    uint4 u = *(const uint4*)(p + off);
    const unsigned short* s = (const unsigned short*)&u;
#pragma unroll
    for (int j = 0; j < 8; ++j) dst[j] = bf2f(s[j]);
  };
  ld8(Q1, r0, q1a); ld8(Q1, r1, q1b);
  ld8(K1, r0, k1a); ld8(K1, r1, k1b);
  ld8(Q2, r0, q2a); ld8(Q2, r1, q2b);
  ld8(K2, r0, k2a); ld8(K2, r1, k2b);
  ld8(V,  r0, va);  ld8(V,  r1, vb);

  float p[8];
#pragma unroll
  for (int i = 0; i < 8; ++i) p[i] = 0.f;
#pragma unroll
  for (int j = 0; j < 8; ++j) {
    p[0] += q1a[j] * k1a[j];  p[1] += q1a[j] * k1b[j];
    p[2] += q1b[j] * k1a[j];  p[3] += q1b[j] * k1b[j];
    p[4] += q2a[j] * k2a[j];  p[5] += q2a[j] * k2b[j];
    p[6] += q2b[j] * k2a[j];  p[7] += q2b[j] * k2b[j];
  }
#pragma unroll
  for (int i = 0; i < 8; ++i) p[i] = wave_sum(p[i]);

  const float l = lam[b];
  float d[2][2];
#pragma unroll
  for (int q = 0; q < 2; ++q) {
    float s0 = p[2 * q]     * SCALE_F, s1 = p[2 * q + 1]     * SCALE_F;
    float t0 = p[4 + 2 * q] * SCALE_F, t1 = p[4 + 2 * q + 1] * SCALE_F;
    float m1 = fmaxf(s0, s1), m2 = fmaxf(t0, t1);
    float e0 = __expf(s0 - m1), e1 = __expf(s1 - m1);
    float f0 = __expf(t0 - m2), f1 = __expf(t1 - m2);
    float i1 = 1.f / (e0 + e1), i2 = 1.f / (f0 + f1);
    d[q][0] = fmaxf(0.f, e0 * i1 - l * f0 * i2);
    d[q][1] = fmaxf(0.f, e1 * i1 - l * f1 * i2);
  }

  union { uint4 u; unsigned short s[8]; } o0, o1;
#pragma unroll
  for (int j = 0; j < 8; ++j) {
    o0.s[j] = f2bf(d[0][0] * va[j] + d[0][1] * vb[j]);
    o1.s[j] = f2bf(d[1][0] * va[j] + d[1][1] * vb[j]);
  }
  *(uint4*)(Aout + r0) = o0.u;
  *(uint4*)(Aout + r1) = o1.u;
}

// ---------- launch ----------
extern "C" void kernel_launch(void* const* d_in, const int* in_sizes, int n_in,
                              void* d_out, int out_size, void* d_ws, size_t ws_size,
                              hipStream_t stream) {
  (void)in_sizes; (void)n_in; (void)out_size; (void)ws_size;
  const float* x    = (const float*)d_in[0];
  const float* WQ1w = (const float*)d_in[1];  const float* WQ1b = (const float*)d_in[2];
  const float* WK1w = (const float*)d_in[3];  const float* WK1b = (const float*)d_in[4];
  const float* WQ2w = (const float*)d_in[5];  const float* WQ2b = (const float*)d_in[6];
  const float* WK2w = (const float*)d_in[7];  const float* WK2b = (const float*)d_in[8];
  const float* WVw  = (const float*)d_in[9];  const float* WVb  = (const float*)d_in[10];
  const float* WOw  = (const float*)d_in[11]; const float* WOb  = (const float*)d_in[12];
  const float* L1w  = (const float*)d_in[13]; const float* L1b  = (const float*)d_in[14];
  const float* L2w  = (const float*)d_in[15]; const float* L2b  = (const float*)d_in[16];

  char* ws = (char*)d_ws;
  size_t off = 0;
  auto take = [&](size_t bytes) {
    char* p = ws + off;
    off = (off + bytes + 255) & ~(size_t)255;
    return p;
  };
  unsigned short* xb   = (unsigned short*)take((size_t)B_SZ * DIM * 2);     // x as bf16
  unsigned short* wq1p = (unsigned short*)take((size_t)SD * SD * 2);
  unsigned short* wk1p = (unsigned short*)take((size_t)SD * SD * 2);
  unsigned short* wq2p = (unsigned short*)take((size_t)SD * SD * 2);
  unsigned short* wk2p = (unsigned short*)take((size_t)SD * SD * 2);
  unsigned short* wvp  = (unsigned short*)take((size_t)SD * SD * 2);
  unsigned short* wop  = (unsigned short*)take((size_t)SD * SD * 2);
  unsigned short* l1p  = (unsigned short*)take((size_t)DIM * LH * 2);
  unsigned short* q1   = (unsigned short*)take((size_t)TOKROWS * SD * 2);
  unsigned short* k1   = (unsigned short*)take((size_t)TOKROWS * SD * 2);
  unsigned short* q2   = (unsigned short*)take((size_t)TOKROWS * SD * 2);
  unsigned short* k2   = (unsigned short*)take((size_t)TOKROWS * SD * 2);
  unsigned short* vv   = (unsigned short*)take((size_t)TOKROWS * SD * 2);
  unsigned short* hbuf = (unsigned short*)take((size_t)B_SZ * LH * 2);
  float*          lam  = (float*)take((size_t)B_SZ * 4);
  unsigned short* aout = (unsigned short*)take((size_t)TOKROWS * SD * 2);

  const int T = 256;
  auto launch_gemm = [&](const unsigned short* A, const unsigned short* Wp, const float* bias,
                         const float* resid, void* out, int M, int N, int K, int obf, int relu) {
    int waves = (M / 64) * (N / 32);
    int blocks = (waves * 32 + T - 1) / T;
    gemm_bf16_wmma<<<blocks, T, 0, stream>>>(A, Wp, bias, resid, out, M, N, K, obf, relu);
  };

  // 1) convert x (f32 -> bf16)
  size_t nx4 = (size_t)B_SZ * DIM / 4;
  cvt_f32_bf16<<<(int)((nx4 + T - 1) / T), T, 0, stream>>>(x, xb, nx4);

  // 2) pack weights into WMMA B-fragment order
  pack_weight<<<(SD * SD + T - 1) / T, T, 0, stream>>>(WQ1w, wq1p, SD, SD);
  pack_weight<<<(SD * SD + T - 1) / T, T, 0, stream>>>(WK1w, wk1p, SD, SD);
  pack_weight<<<(SD * SD + T - 1) / T, T, 0, stream>>>(WQ2w, wq2p, SD, SD);
  pack_weight<<<(SD * SD + T - 1) / T, T, 0, stream>>>(WK2w, wk2p, SD, SD);
  pack_weight<<<(SD * SD + T - 1) / T, T, 0, stream>>>(WVw,  wvp,  SD, SD);
  pack_weight<<<(SD * SD + T - 1) / T, T, 0, stream>>>(WOw,  wop,  SD, SD);
  pack_weight<<<(DIM * LH + T - 1) / T, T, 0, stream>>>(L1w, l1p, DIM, LH);

  // 3) five projection GEMMs: [32768,1024] x [1024,1024] -> bf16
  launch_gemm(xb, wq1p, WQ1b, nullptr, q1, TOKROWS, SD, SD, 1, 0);
  launch_gemm(xb, wk1p, WK1b, nullptr, k1, TOKROWS, SD, SD, 1, 0);
  launch_gemm(xb, wq2p, WQ2b, nullptr, q2, TOKROWS, SD, SD, 1, 0);
  launch_gemm(xb, wk2p, WK2b, nullptr, k2, TOKROWS, SD, SD, 1, 0);
  launch_gemm(xb, wvp,  WVb,  nullptr, vv, TOKROWS, SD, SD, 1, 0);

  // 4) lambda hidden: relu(x @ L1 + b) -> bf16 [16384,256]
  launch_gemm(xb, l1p, L1b, nullptr, hbuf, B_SZ, LH, DIM, 1, 1);

  // 5) lambda finish (sigmoid of 256-dot)
  lambda_final<<<(B_SZ * 32) / T, T, 0, stream>>>(hbuf, L2w, L2b, lam, B_SZ);

  // 6) differential attention, one wave per (b,h)
  diff_attn_kernel<<<(B_SZ * NHEADS * 32) / T, T, 0, stream>>>(q1, k1, q2, k2, vv, lam, aout);

  // 7) output projection + residual (residual = tokens = x viewed [32768,1024]) -> f32 d_out
  launch_gemm(aout, wop, WOb, x, d_out, TOKROWS, SD, SD, 0, 0);
}
